// ReduceContextAttentionP2_28698971472633
// MI455X (gfx1250) — compile-verified, matrix operand order checked by reading, not run on hardware
//
#include <hip/hip_runtime.h>

// ---------------------------------------------------------------------------
// MI455X (gfx1250) implementation.  All heavy math as f16 WMMA GEMMs with f32
// accumulate; B operands pre-transposed to [N][K] so both LDS tiles are
// K-contiguous and every WMMA fragment loads as two ds_load_b128.
//
//   BkT [ 768][3976] f16 = im2col(edge-pad(b))^T
//   MkT [ 256][23816]f16 = im2col(edge-pad(mask))^T
//   AkT [4608][3976] f16 = im2col(edge-pad(aux))^T
//   St  [3969][5768] f16 = cos_similar^T
//   C1  = conv1_w @ BkT^T + b1      C2 = conv2_w @ MkT^T + b2
//   C3  = convaux_w @ AkT^T + ba
//   C1mT[768][5768] f16 = (C1*(1-bcast C2))^T ;  C2hT[256][5768] f16 = C2^T
//   P1 = St @ C1mT^T ;  P2 = St @ C2hT^T
//   output = crop4(fold(P1)); hole = crop4(up2(fold(P2))/up3(weight_map));
//   raw = crop4(fold(gather(C3, argmax))/weight_map)
// ---------------------------------------------------------------------------

typedef __attribute__((ext_vector_type(16))) _Float16 v16h;
typedef __attribute__((ext_vector_type(8)))  _Float16 v8h;
typedef __attribute__((ext_vector_type(8)))  float    v8f;

#define LP      3969
#define NP      63
#define CHN     5766
#define IMG     504
#define PAD     4
#define PIMG    512
#define LDB_BK  3976     // 3969 padded to mult-of-8 halves (16B rows)
#define LDB_MK  23816    // 23814 padded
#define LDB_ST  5768     // 5766 padded

// patch coverage range for padded coordinate v in [0,512)
__device__ __forceinline__ int cov_lo(int v) { return v >= 15 ? ((v - 8) >> 3) : 0; }
__device__ __forceinline__ int cov_hi(int v) { int h = v >> 3; return h > 62 ? 62 : h; }

// CDNA5 async global->LDS copy (ASYNCcnt tracked) ---------------------------
__device__ __forceinline__ void async_g2l_b128(_Float16* lds, const _Float16* g) {
  unsigned l32 = (unsigned)(unsigned long long)(void*)lds;  // low 32 = LDS offset
  asm volatile("global_load_async_to_lds_b128 %0, %1, off"
               :: "v"(l32), "v"(g) : "memory");
}
__device__ __forceinline__ void wait_async0() {
#if __has_builtin(__builtin_amdgcn_s_wait_asynccnt)
  __builtin_amdgcn_s_wait_asynccnt(0);
#else
  asm volatile("s_wait_asynccnt 0x0" ::: "memory");
#endif
}

// ---------------- im2col, transposed output: out[n][k] ---------------------
// mode 0 (b/aux):  K=LP     k=l,        n=c*256+p
// mode 1 (mask):   K=LP*6   k=l*6+c,    n=p
__global__ __launch_bounds__(256) void im2colT_kernel(
    const float* __restrict__ img, _Float16* __restrict__ out,
    int C, int ldbT, int mode, long total)
{
  long i = (long)blockIdx.x * blockDim.x + threadIdx.x;
  if (i >= total) return;
  int n, l, c, p, k;
  if (mode == 0) {
    k = (int)(i % LP);  n = (int)(i / LP);
    l = k;  c = n >> 8;  p = n & 255;
  } else {
    k = (int)(i % (LP * 6));  n = (int)(i / (LP * 6));
    l = k / 6;  c = k % 6;    p = n;
  }
  int ky = p >> 4, kx = p & 15;
  int ph = l / NP, pw = l % NP;
  int y = ph * 8 + ky - PAD;
  int x = pw * 8 + kx - PAD;
  y = y < 0 ? 0 : (y > IMG - 1 ? IMG - 1 : y);
  x = x < 0 ? 0 : (x > IMG - 1 ? IMG - 1 : x);
  out[(size_t)n * ldbT + k] = (_Float16)img[((size_t)c * IMG + y) * IMG + x];
}

// ---------- transpose cos_similar [CH][LP] -> St [LP][LDB_ST] f16 ----------
__global__ __launch_bounds__(256) void transpose_f16_kernel(
    const float* __restrict__ S, _Float16* __restrict__ St)
{
  __shared__ float tile[32][33];
  int bx = blockIdx.x * 32;   // patch l
  int by = blockIdx.y * 32;   // channel
  int tx = threadIdx.x, ty = threadIdx.y;   // (32, 8)
#pragma unroll
  for (int i = 0; i < 32; i += 8) {
    int r = by + ty + i, c = bx + tx;
    tile[ty + i][tx] = (r < CHN && c < LP) ? S[(size_t)r * LP + c] : 0.f;
  }
  __syncthreads();
#pragma unroll
  for (int i = 0; i < 32; i += 8) {
    int l = bx + ty + i, ch = by + tx;
    if (l < LP && ch < CHN) St[(size_t)l * LDB_ST + ch] = (_Float16)tile[tx][ty + i];
  }
}

// ------------------------------- argmax ------------------------------------
__global__ __launch_bounds__(256) void argmax_kernel(
    const float* __restrict__ S, int* __restrict__ idx)
{
  int l = blockIdx.x * blockDim.x + threadIdx.x;
  if (l >= LP) return;
  float best = S[l];
  int   bi   = 0;
  for (int c = 1; c < CHN; ++c) {
    float v = S[(size_t)c * LP + l];
    if (v > best) { best = v; bi = c; }
  }
  idx[l] = bi;
}

// --------------------------- WMMA tiled GEMM -------------------------------
// C[M,N] f32 = A[M,K] (f32 or f16, row-major lda) x BT[N,K]^T (f16, ldbT) + bias
// Block 128x128x32, 8 waves (2x4), wave tile 64x32 -> 8 v_wmma per k-step.
// Double-buffered LDS; B (and f16 A) staged with async global->LDS.
template <typename AT>
__global__ __launch_bounds__(256) void gemm_wmma_kernel(
    const AT* __restrict__ A, int lda,
    const _Float16* __restrict__ BT, int ldbT,
    float* __restrict__ C, int ldc,
    const float* __restrict__ bias,
    int M, int N, int K)
{
  constexpr int BM = 128, BN = 128, BK = 32, LDT = 40;  // padded LDS row stride
  __shared__ _Float16 As[2][BM * LDT];
  __shared__ _Float16 Bs[2][BN * LDT];

  const int tid  = threadIdx.x;
  const int wave = tid >> 5;
  const int lane = tid & 31;
  const int m0 = blockIdx.y * BM;
  const int n0 = blockIdx.x * BN;

  const int wm0 = (wave >> 2) * 64;
  const int wn0 = (wave & 3) * 32;
  const int halfsel = lane >> 4;
  const int lm = lane & 15;

  const int arow = tid >> 1;            // 0..127
  const int acol = (tid & 1) * 16;      // 0 / 16

  v8f acc[4][2];
#pragma unroll
  for (int mi = 0; mi < 4; ++mi)
#pragma unroll
    for (int nj = 0; nj < 2; ++nj)
      acc[mi][nj] = (v8f){0.f, 0.f, 0.f, 0.f, 0.f, 0.f, 0.f, 0.f};

  float ar[16];          // staging regs for f32-A fast path
  bool  adirect = false; // true if issue already wrote As (tail / f16 / OOB)

  auto issue = [&](int k0, int bi) {
    const bool fullK = (k0 + BK <= K);
    // ---- B tile: rows = n (always in range; N is a multiple of 128) ----
    {
      _Float16*       bl = &Bs[bi][arow * LDT + acol];
      const _Float16* bg = BT + (size_t)(n0 + arow) * ldbT + k0 + acol;
      if (fullK) {
        async_g2l_b128(bl, bg);
        async_g2l_b128(bl + 8, bg + 8);
      } else {
#pragma unroll
        for (int u = 0; u < 16; ++u) {
          int gk = k0 + acol + u;
          bl[u] = (gk < K) ? bg[u] : (_Float16)0.f;
        }
      }
    }
    // ---- A tile ----
    const int  gm  = m0 + arow;
    const bool mok = gm < M;
    _Float16*  al  = &As[bi][arow * LDT + acol];
    const AT*  ag  = A + (size_t)gm * lda + k0 + acol;
    adirect = false;
    if constexpr (sizeof(AT) == 2) {
      if (fullK && mok) {
        async_g2l_b128(al, (const _Float16*)ag);
        async_g2l_b128(al + 8, (const _Float16*)ag + 8);
      } else {
#pragma unroll
        for (int u = 0; u < 16; ++u) {
          int gk = k0 + acol + u;
          al[u] = (mok && gk < K) ? (_Float16)ag[u] : (_Float16)0.f;
        }
      }
      adirect = true;
    } else {
      if (fullK && mok) {
#pragma unroll
        for (int u = 0; u < 16; ++u) ar[u] = (float)ag[u];
        if (k0 + 2 * BK <= K) __builtin_prefetch(ag + BK, 0, 1);
      } else {
#pragma unroll
        for (int u = 0; u < 16; ++u) {
          int gk = k0 + acol + u;
          float v = (mok && gk < K) ? (float)ag[u] : 0.f;
          al[u] = (_Float16)v;
        }
        adirect = true;
      }
    }
  };

  auto commit = [&](int bi) {
    if constexpr (sizeof(AT) == 4) {
      if (!adirect) {
        v8h lo, hi;
#pragma unroll
        for (int u = 0; u < 8; ++u) {
          lo[u] = (_Float16)ar[u];
          hi[u] = (_Float16)ar[u + 8];
        }
        *(v8h*)&As[bi][arow * LDT + acol]     = lo;
        *(v8h*)&As[bi][arow * LDT + acol + 8] = hi;
      }
    }
  };

  auto compute = [&](int bi) {
    const _Float16* Asb = &As[bi][0];
    const _Float16* Bsb = &Bs[bi][0];
    v16h bf[2];
#pragma unroll
    for (int nj = 0; nj < 2; ++nj) {
      const _Float16* bp = Bsb + (wn0 + nj * 16 + lm) * LDT + halfsel * 8;
      v8h b0 = *(const v8h*)bp;          // k = halfsel*8 + 0..7
      v8h b1 = *(const v8h*)(bp + 16);   // k = 16 + halfsel*8 + 0..7
      bf[nj] = __builtin_shufflevector(b0, b1, 0, 1, 2, 3, 4, 5, 6, 7,
                                       8, 9, 10, 11, 12, 13, 14, 15);
    }
#pragma unroll
    for (int mi = 0; mi < 4; ++mi) {
      const _Float16* ap = Asb + (wm0 + mi * 16 + lm) * LDT + halfsel * 8;
      v8h a0 = *(const v8h*)ap;
      v8h a1 = *(const v8h*)(ap + 16);
      v16h af = __builtin_shufflevector(a0, a1, 0, 1, 2, 3, 4, 5, 6, 7,
                                        8, 9, 10, 11, 12, 13, 14, 15);
#pragma unroll
      for (int nj = 0; nj < 2; ++nj) {
        acc[mi][nj] = __builtin_amdgcn_wmma_f32_16x16x32_f16(
            false, af, false, bf[nj], (short)0, acc[mi][nj], false, false);
      }
    }
  };

  const int nk = (K + BK - 1) / BK;
  issue(0, 0);
  commit(0);
  wait_async0();
  __syncthreads();

  for (int kt = 0; kt < nk; ++kt) {
    const int  cur  = kt & 1;
    const bool more = (kt + 1 < nk);
    if (more) issue((kt + 1) * BK, cur ^ 1);
    compute(cur);
    if (more) commit(cur ^ 1);
    wait_async0();
    __syncthreads();
  }

  // epilogue (N always in-bounds; guard M only)
  const float* bp = bias;
#pragma unroll
  for (int mi = 0; mi < 4; ++mi)
#pragma unroll
    for (int r = 0; r < 8; ++r) {
      int m = m0 + wm0 + mi * 16 + r + halfsel * 8;
      if (m < M) {
        float bv = bp ? bp[m] : 0.f;
#pragma unroll
        for (int nj = 0; nj < 2; ++nj) {
          int n = n0 + wn0 + nj * 16 + lm;
          C[(size_t)m * ldc + n] = acc[mi][nj][r] + bv;
        }
      }
    }
}

// -------- combine: C1mT[j][m] = C1*(1-bcast C2); C2hT[p][m] = C2 -----------
__global__ __launch_bounds__(256) void combine_kernel(
    const float* __restrict__ C1, const float* __restrict__ C2,
    _Float16* __restrict__ C1mT, _Float16* __restrict__ C2hT)
{
  long i = (long)blockIdx.x * blockDim.x + threadIdx.x;
  if (i >= (long)768 * CHN) return;
  int j = (int)(i / CHN);   // 0..767
  int m = (int)(i % CHN);   // coalesced writes
  float ms = C2[(size_t)m * 256 + (j & 255)];
  C1mT[(size_t)j * LDB_ST + m] = (_Float16)(C1[(size_t)m * 768 + j] * (1.0f - ms));
  if (j < 256) C2hT[(size_t)j * LDB_ST + m] = (_Float16)ms;
}

// ----------------- fold (overlap-add) of P1 -> output (crop 4) -------------
__global__ __launch_bounds__(256) void fold_output_kernel(
    const float* __restrict__ P1, float* __restrict__ out)
{
  int i = blockIdx.x * blockDim.x + threadIdx.x;
  if (i >= 3 * IMG * IMG) return;
  int x = i % IMG; int t = i / IMG; int y = t % IMG; int c = t / IMG;
  int yp = y + PAD, xp = x + PAD;
  int yl = cov_lo(yp), yh = cov_hi(yp);
  int xl = cov_lo(xp), xh = cov_hi(xp);
  float s = 0.f;
  for (int ph = yl; ph <= yh; ++ph)
    for (int pw = xl; pw <= xh; ++pw) {
      int l = ph * NP + pw;
      s += P1[(size_t)l * 768 + c * 256 + (yp - 8 * ph) * 16 + (xp - 8 * pw)];
    }
  out[i] = s;
}

// ----------------- fold of P2 -> mask_recon image (512x512) ----------------
__global__ __launch_bounds__(256) void fold_mr_kernel(
    const float* __restrict__ P2, float* __restrict__ mr)
{
  int i = blockIdx.x * blockDim.x + threadIdx.x;
  if (i >= PIMG * PIMG) return;
  int x = i % PIMG, y = i / PIMG;
  int yl = cov_lo(y), yh = cov_hi(y);
  int xl = cov_lo(x), xh = cov_hi(x);
  float s = 0.f;
  for (int ph = yl; ph <= yh; ++ph)
    for (int pw = xl; pw <= xh; ++pw) {
      int l = ph * NP + pw;
      s += P2[(size_t)l * 256 + (y - 8 * ph) * 16 + (x - 8 * pw)];
    }
  mr[i] = s;
}

// ----------------- hole = crop4( up2(mr) / up3(weight_map) ) ---------------
__global__ __launch_bounds__(256) void hole_kernel(
    const float* __restrict__ mr,
    const float* __restrict__ up2w, const float* __restrict__ up2b,
    const float* __restrict__ up3w, const float* __restrict__ up3b,
    float* __restrict__ out)
{
  long i = (long)blockIdx.x * blockDim.x + threadIdx.x;
  const long total = 6L * 1016 * 1016;
  if (i >= total) return;
  int X = (int)(i % 1016); long t = i / 1016;
  int Y = (int)(t % 1016); int c = (int)(t / 1016);
  int Yp = Y + PAD, Xp = X + PAD;
  int yy = Yp >> 1, xx = Xp >> 1, py = Yp & 1, px = Xp & 1;
  float wmv = (float)((cov_hi(yy) - cov_lo(yy) + 1) * (cov_hi(xx) - cov_lo(xx) + 1));
  float num = mr[yy * PIMG + xx] * up2w[c * 4 + py * 2 + px] + up2b[c];
  float den = wmv * up3w[py * 2 + px] + up3b[0];
  out[i] = num / den;
}

// ------- raw = crop4( fold(gather(C3, argmax)) / weight_map ) --------------
__global__ __launch_bounds__(256) void fold_raw_kernel(
    const float* __restrict__ C3, const int* __restrict__ idx,
    float* __restrict__ out)
{
  long i = (long)blockIdx.x * blockDim.x + threadIdx.x;
  const long total = 18L * IMG * IMG;
  if (i >= total) return;
  int x = (int)(i % IMG); long t = i / IMG;
  int y = (int)(t % IMG); int f = (int)(t / IMG);
  int yp = y + PAD, xp = x + PAD;
  int yl = cov_lo(yp), yh = cov_hi(yp);
  int xl = cov_lo(xp), xh = cov_hi(xp);
  float s = 0.f;
  for (int ph = yl; ph <= yh; ++ph)
    for (int pw = xl; pw <= xh; ++pw) {
      int l  = ph * NP + pw;
      int ch = idx[l];
      s += C3[(size_t)ch * 4608 + f * 256 + (yp - 8 * ph) * 16 + (xp - 8 * pw)];
    }
  float wmv = (float)((yh - yl + 1) * (xh - xl + 1));
  out[i] = s / wmv;
}

// ---------------------------------------------------------------------------
extern "C" void kernel_launch(void* const* d_in, const int* in_sizes, int n_in,
                              void* d_out, int out_size, void* d_ws, size_t ws_size,
                              hipStream_t stream) {
  const float* cosS = (const float*)d_in[0];
  const float* bimg = (const float*)d_in[1];
  const float* mask = (const float*)d_in[2];
  const float* aux  = (const float*)d_in[3];
  const float* w1   = (const float*)d_in[4];
  const float* b1   = (const float*)d_in[5];
  const float* w2   = (const float*)d_in[6];
  const float* b2   = (const float*)d_in[7];
  const float* wa   = (const float*)d_in[8];
  const float* ba   = (const float*)d_in[9];
  const float* up2w = (const float*)d_in[10];
  const float* up2b = (const float*)d_in[11];
  const float* up3w = (const float*)d_in[12];
  const float* up3b = (const float*)d_in[13];
  float* out = (float*)d_out;

  char* ws = (char*)d_ws;
  auto alloc = [&](size_t bytes) -> char* {
    char* p = ws;
    ws += (bytes + 255) & ~(size_t)255;
    return p;
  };
  _Float16* BkT  = (_Float16*)alloc((size_t)768 * LDB_BK * 2);
  _Float16* MkT  = (_Float16*)alloc((size_t)256 * LDB_MK * 2);
  _Float16* AkT  = (_Float16*)alloc((size_t)4608 * LDB_BK * 2);
  _Float16* St   = (_Float16*)alloc((size_t)LP * LDB_ST * 2);
  float*    C1   = (float*)alloc((size_t)CHN * 768 * 4);
  float*    C2   = (float*)alloc((size_t)CHN * 256 * 4);
  float*    C3   = (float*)alloc((size_t)CHN * 4608 * 4);
  _Float16* C1mT = (_Float16*)alloc((size_t)768 * LDB_ST * 2);
  _Float16* C2hT = (_Float16*)alloc((size_t)256 * LDB_ST * 2);
  float*    P1   = (float*)alloc((size_t)LP * 768 * 4);
  float*    P2   = (float*)alloc((size_t)LP * 256 * 4);
  float*    mr   = (float*)alloc((size_t)PIMG * PIMG * 4);
  int*      idx  = (int*)alloc((size_t)LP * 4);

  auto blocks = [](long n) { return (unsigned)((n + 255) / 256); };

  // 1) im2col (transposed outputs)
  im2colT_kernel<<<blocks(768L * LP), 256, 0, stream>>>(bimg, BkT, 3, LDB_BK, 0, 768L * LP);
  im2colT_kernel<<<blocks(256L * LP * 6), 256, 0, stream>>>(mask, MkT, 6, LDB_MK, 1, 256L * LP * 6);
  im2colT_kernel<<<blocks(4608L * LP), 256, 0, stream>>>(aux, AkT, 18, LDB_BK, 0, 4608L * LP);

  // 2) St = cos_similar^T (f16);  argmax per patch
  {
    dim3 g((LP + 31) / 32, (CHN + 31) / 32), b(32, 8);
    transpose_f16_kernel<<<g, b, 0, stream>>>(cosS, St);
  }
  argmax_kernel<<<blocks(LP), 256, 0, stream>>>(cosS, idx);

  // 3) kernel-generation GEMMs (A = f32 weights)
  {
    dim3 g(768 / 128, (CHN + 127) / 128);
    gemm_wmma_kernel<float><<<g, 256, 0, stream>>>(w1, LP, BkT, LDB_BK, C1, 768, b1, CHN, 768, LP);
  }
  {
    dim3 g(256 / 128, (CHN + 127) / 128);
    gemm_wmma_kernel<float><<<g, 256, 0, stream>>>(w2, LP * 6, MkT, LDB_MK, C2, 256, b2, CHN, 256, LP * 6);
  }
  {
    dim3 g(4608 / 128, (CHN + 127) / 128);
    gemm_wmma_kernel<float><<<g, 256, 0, stream>>>(wa, LP, AkT, LDB_BK, C3, 4608, ba, CHN, 4608, LP);
  }

  // 4) bkg*(1-msk) + f16 transposed copies
  combine_kernel<<<blocks(768L * CHN), 256, 0, stream>>>(C1, C2, C1mT, C2hT);

  // 5) transposed-conv GEMMs (A = St f16, async-staged)
  {
    dim3 g(768 / 128, (LP + 127) / 128);
    gemm_wmma_kernel<_Float16><<<g, 256, 0, stream>>>(St, LDB_ST, C1mT, LDB_ST, P1, 768,
                                                      (const float*)nullptr, LP, 768, CHN);
  }
  {
    dim3 g(256 / 128, (LP + 127) / 128);
    gemm_wmma_kernel<_Float16><<<g, 256, 0, stream>>>(St, LDB_ST, C2hT, LDB_ST, P2, 256,
                                                      (const float*)nullptr, LP, 256, CHN);
  }

  // 6) folds + epilogues -> outputs (concat order: output, hole, raw)
  float* out_output = out;
  float* out_hole   = out + (size_t)3 * IMG * IMG;
  float* out_raw    = out_hole + (size_t)6 * 1016 * 1016;

  fold_output_kernel<<<blocks(3L * IMG * IMG), 256, 0, stream>>>(P1, out_output);
  fold_mr_kernel<<<blocks((long)PIMG * PIMG), 256, 0, stream>>>(P2, mr);
  hole_kernel<<<blocks(6L * 1016 * 1016), 256, 0, stream>>>(mr, up2w, up2b, up3w, up3b, out_hole);
  fold_raw_kernel<<<blocks(18L * IMG * IMG), 256, 0, stream>>>(C3, idx, out_raw);
}
